// GCFNet_35536559407119
// MI455X (gfx1250) — matrix operand order repaired and checked by reading, not compiled, not run on hardware
//
#include <hip/hip_runtime.h>
#include <cstdint>
#include <cstddef>

// ---------------------------------------------------------------------------
// GCFNet forward for MI455X (gfx1250, wave32, WMMA).
// - every dense layer: f16 v_wmma_f32_16x16x32_f16 GEMM, f32 accumulation
// - GEMM B tiles staged to LDS by the Tensor Data Mover (tensor_load_to_lds,
//   TENSORcnt), wave-cooperative fallback under GEMM_USE_TDM=0
// - residual/LN stream kept f32; all GEMM operands stored in the row-major
//   layouts that make CDNA5 A/B fragments contiguous loads
// ---------------------------------------------------------------------------

typedef __attribute__((ext_vector_type(16))) _Float16 v16h;
typedef __attribute__((ext_vector_type(8)))  _Float16 v8h;
typedef __attribute__((ext_vector_type(8)))  float    v8f;
typedef __attribute__((ext_vector_type(4)))  unsigned int u32x4;
typedef __attribute__((ext_vector_type(8)))  unsigned int u32x8;

#define NN    2048
#define TT    64
#define DIN   16
#define HH    128
#define CTXC  32
#define CEC   32
#define NHD   4
#define DHD   32
#define NCHAIN 200
#define NIND1  30
#define NIND2  100

#ifndef GEMM_USE_TDM
#define GEMM_USE_TDM 1   // set to 0 to fall back to cooperative LDS staging
#endif

enum { ACT_NONE = 0, ACT_RELU = 1, ACT_TANH = 2, ACT_SILU = 3, ACT_SIGMOID = 4 };

// ---------------- device helpers ----------------

__device__ __forceinline__ int clampi(int v, int hi) {
  return v < 0 ? 0 : (v > hi ? hi : v);
}

// A-fragment in this code's k order (B uses the same) for V_WMMA_F32_16X16X32_F16 (16x32 f16, M x K):
// lane l (row = l&15, half = l>>4) holds K = half*16..half*16+15
// -> two contiguous 16B chunks of a row-major row.
__device__ __forceinline__ v16h load_afrag(const _Float16* rowp, int k0, int half) {
  v8h lo = *reinterpret_cast<const v8h*>(rowp + k0 + half * 16);
  v8h hi = *reinterpret_cast<const v8h*>(rowp + k0 + half * 16 + 8);
  v16h r;
#pragma unroll
  for (int i = 0; i < 8; ++i) { r[i] = lo[i]; r[i + 8] = hi[i]; }
  return r;
}

template <int ACT> __device__ __forceinline__ float act_apply(float v) {
  if constexpr (ACT == ACT_RELU)         return v > 0.f ? v : 0.f;
  else if constexpr (ACT == ACT_TANH)    return tanhf(v);
  else if constexpr (ACT == ACT_SILU)    return v / (1.f + expf(-v));
  else if constexpr (ACT == ACT_SIGMOID) return 1.f / (1.f + expf(-v));
  else return v;
}

// Tensor Data Mover: DMA a [64 rows x 32 elem] f16 tile (row stride = K elems)
// from global into LDS, rows compacted contiguously (64B per row) -- exactly
// the B-fragment layout used below.  D# built per CDNA5 ISA ch.8.
__device__ __forceinline__ void tdm_load_b_tile(const _Float16* gsrc,
                                                unsigned ldsByte, int K) {
  const unsigned long long ga = (unsigned long long)gsrc;
  u32x4 g0;
  g0[0] = 1u;                                            // count=1, user mode
  g0[1] = ldsByte;                                       // lds_addr[31:0]
  g0[2] = (unsigned)ga;                                  // global_addr[31:0]
  g0[3] = (unsigned)((ga >> 32) & 0x01FFFFFFull)         // global_addr[56:32]
        | (2u << 30);                                    // type=2 ("image")
  u32x8 g1;
  g1[0] = 1u << 16;                                      // data_size=1 (2 bytes)
  g1[1] = ((unsigned)K & 0xFFFFu) << 16;                 // tensor_dim0[15:0]
  g1[2] = (((unsigned)K >> 16) & 0xFFFFu)                // tensor_dim0[31:16]
        | (64u << 16);                                   // tensor_dim1[15:0]=64
  g1[3] = (32u << 16);                                   // tile_dim0=32
  g1[4] = 64u;                                           // tile_dim1=64, tile_dim2=0
  g1[5] = (unsigned)K;                                   // tensor_dim0_stride[31:0]
  g1[6] = 0u;                                            // stride hi / dim1_stride lo
  g1[7] = 0u;
  asm volatile("tensor_load_to_lds %0, %1" :: "s"(g0), "s"(g1) : "memory");
}

// ---------------- generic WMMA GEMM:  C[M,N] = act(A[M,K] @ W[N,K]^T + bias) --
// Block: 128 threads = 4 waves; tile 128(M) x 64(N); K stepped by 32.
// Each wave owns 32 rows (2 A-frags, 8 accumulators) -> 8 WMMA per K-step.
// B tile (64 cols x 32 K = 4KB f16) staged into LDS via TDM (wave 0 issues,
// waits TENSORcnt, workgroup barrier publishes) or cooperatively.

template <int ACT, bool OUT16>
__global__ void __launch_bounds__(128) gemm_wmma_kernel(
    const _Float16* __restrict__ A, const _Float16* __restrict__ W,
    const float* __restrict__ bias, float* __restrict__ Cf,
    _Float16* __restrict__ Ch, int M, int Nout, int K) {
  __shared__ __align__(64) _Float16 ldsB[64 * 32];
  const int bm = blockIdx.x, bn = blockIdx.y;
  const int tid = threadIdx.x;
  const int wave = tid >> 5, lane = tid & 31, half = lane >> 4, l15 = lane & 15;
  const _Float16* Arow0 = A + (size_t)(bm * 128 + wave * 32 + l15) * K;
  const _Float16* Arow1 = Arow0 + (size_t)16 * K;
  v8f acc0[4] = {};
  v8f acc1[4] = {};
  for (int k0 = 0; k0 < K; k0 += 32) {
    __syncthreads();  // previous iteration's B reads complete
#if GEMM_USE_TDM
    if (wave == 0) {
      tdm_load_b_tile(W + (size_t)(bn * 64) * K + k0,
                      (unsigned)(unsigned long long)(&ldsB[0]), K);
      __builtin_amdgcn_s_wait_tensorcnt(0);
    }
#else
    {
      const int col = tid >> 1, h2 = tid & 1;
      const _Float16* src = W + (size_t)(bn * 64 + col) * K + k0 + h2 * 16;
      v8h* dst = reinterpret_cast<v8h*>(&ldsB[(col * 2 + h2) * 16]);
      dst[0] = *reinterpret_cast<const v8h*>(src);
      dst[1] = *reinterpret_cast<const v8h*>(src + 8);
      if (k0 + 32 < K) __builtin_prefetch((const void*)(src + 32), 0, 0);
    }
#endif
    __syncthreads();  // B tile visible to all waves
    const v16h a0 = load_afrag(Arow0, k0, half);
    const v16h a1 = load_afrag(Arow1, k0, half);
#pragma unroll
    for (int c = 0; c < 4; ++c) {
      const v16h b =
          *reinterpret_cast<const v16h*>(&ldsB[((c * 16 + l15) * 2 + half) * 16]);
      acc0[c] = __builtin_amdgcn_wmma_f32_16x16x32_f16(false, a0, false, b,
                                                       (short)0, acc0[c], false, false);
      acc1[c] = __builtin_amdgcn_wmma_f32_16x16x32_f16(false, a1, false, b,
                                                       (short)0, acc1[c], false, false);
    }
  }
  // D layout: lanes 0-15 rows 0..7 (vgpr r), lanes 16-31 rows 8..15; col = l15.
#pragma unroll
  for (int c = 0; c < 4; ++c) {
    const int col = bn * 64 + c * 16 + l15;
    const float bv = bias ? bias[col] : 0.f;
#pragma unroll
    for (int r = 0; r < 8; ++r) {
      const int orow = bm * 128 + wave * 32 + 8 * half + r;
      const float v0 = act_apply<ACT>(acc0[c][r] + bv);
      const float v1 = act_apply<ACT>(acc1[c][r] + bv);
      if constexpr (OUT16) {
        Ch[(size_t)orow * Nout + col] = (_Float16)v0;
        Ch[(size_t)(orow + 16) * Nout + col] = (_Float16)v1;
      } else {
        Cf[(size_t)orow * Nout + col] = v0;
        Cf[(size_t)(orow + 16) * Nout + col] = v1;
      }
    }
  }
}

// ---------------- conv + GLU:  x[n,t,h] = a * sigmoid(b) --------------------

__global__ void __launch_bounds__(128) conv_glu_kernel(
    const float* __restrict__ daily, const float* __restrict__ cw,
    const float* __restrict__ cb, float* __restrict__ x32,
    _Float16* __restrict__ x16) {
  const int nt = blockIdx.x;
  const int t = nt % TT, n = nt / TT;
  __shared__ float din[3 * DIN];
  const int tid = threadIdx.x;
  if (tid < 3 * DIN) {
    const int dt = tid / DIN, c = tid % DIN;
    const int tt = t + dt - 1;
    din[tid] = (tt >= 0 && tt < TT) ? daily[((size_t)n * TT + tt) * DIN + c] : 0.f;
  }
  __syncthreads();
  const int h = tid;
  float a = cb[h], b = cb[h + HH];
#pragma unroll
  for (int dt = 0; dt < 3; ++dt)
#pragma unroll
    for (int c = 0; c < DIN; ++c) {
      const float x = din[dt * DIN + c];
      a += cw[((h) * DIN + c) * 3 + dt] * x;
      b += cw[((h + HH) * DIN + c) * 3 + dt] * x;
    }
  const float g = a * (1.f / (1.f + expf(-b)));
  x32[(size_t)nt * HH + h] = g;
  x16[(size_t)nt * HH + h] = (_Float16)g;
}

// ---------------- split qkv into q/k (T x DH) and v^T (DH x T) per head -----

__global__ void split_qkv_kernel(const _Float16* __restrict__ qkv,
                                 _Float16* __restrict__ q, _Float16* __restrict__ k,
                                 _Float16* __restrict__ vT) {
  const size_t i = (size_t)blockIdx.x * 256 + threadIdx.x;
  if (i >= (size_t)NN * TT * HH) return;
  const int h = (int)(i % HH);
  const size_t nt = i / HH;
  const int t = (int)(nt % TT);
  const int n = (int)(nt / TT);
  const int head = h >> 5, d = h & 31;
  const size_t base = (size_t)n * NHD + head;
  q[(base * TT + t) * DHD + d] = qkv[nt * (3 * HH) + h];
  k[(base * TT + t) * DHD + d] = qkv[nt * (3 * HH) + HH + h];
  vT[(base * DHD + d) * TT + t] = qkv[nt * (3 * HH) + 2 * HH + h];
}

// ---------------- fused attention per (asset, head): WMMA + wave32 softmax --

__global__ void __launch_bounds__(128) attn_kernel(
    const _Float16* __restrict__ q, const _Float16* __restrict__ k,
    const _Float16* __restrict__ vT, _Float16* __restrict__ o) {
  __shared__ __align__(64) _Float16 satt[64 * 64];
  const int nh = blockIdx.x;
  const int n = nh / NHD, head = nh % NHD;
  const _Float16* qb = q + (size_t)nh * TT * DHD;
  const _Float16* kb = k + (size_t)nh * TT * DHD;
  const _Float16* vb = vT + (size_t)nh * DHD * TT;
  const int tid = threadIdx.x, wave = tid >> 5, lane = tid & 31;
  const int half = lane >> 4, l15 = lane & 15;

  // scores[m0..m0+15, 0..63] = q @ k^T, K = 32 (one WMMA step)
  v8f accs[4] = {};
  {
    const v16h a = load_afrag(qb + (size_t)(wave * 16 + l15) * DHD, 0, half);
#pragma unroll
    for (int c = 0; c < 4; ++c) {
      const v16h b =
          *reinterpret_cast<const v16h*>(kb + (size_t)(c * 16 + l15) * DHD + half * 16);
      accs[c] = __builtin_amdgcn_wmma_f32_16x16x32_f16(false, a, false, b,
                                                       (short)0, accs[c], false, false);
    }
  }
  // softmax per row: each row lives in 16 lanes (D layout) -> width-16 shfl_xor
  const float scale = 0.17677669529663687f; // 1/sqrt(32)
#pragma unroll
  for (int r = 0; r < 8; ++r) {
    float v0 = accs[0][r] * scale, v1 = accs[1][r] * scale;
    float v2 = accs[2][r] * scale, v3 = accs[3][r] * scale;
    float m = fmaxf(fmaxf(v0, v1), fmaxf(v2, v3));
    for (int off = 8; off >= 1; off >>= 1) m = fmaxf(m, __shfl_xor(m, off, 16));
    const float e0 = expf(v0 - m), e1 = expf(v1 - m), e2 = expf(v2 - m), e3 = expf(v3 - m);
    float s = e0 + e1 + e2 + e3;
    for (int off = 8; off >= 1; off >>= 1) s += __shfl_xor(s, off, 16);
    const float inv = 1.f / s;
    const int orow = wave * 16 + 8 * half + r;
    satt[orow * 64 + 0 * 16 + l15] = (_Float16)(e0 * inv);
    satt[orow * 64 + 1 * 16 + l15] = (_Float16)(e1 * inv);
    satt[orow * 64 + 2 * 16 + l15] = (_Float16)(e2 * inv);
    satt[orow * 64 + 3 * 16 + l15] = (_Float16)(e3 * inv);
  }
  __syncthreads();

  // o = att @ v, K = 64 (two WMMA steps); B frag from v^T (contiguous)
  v8f acco[2] = {};
#pragma unroll
  for (int k0 = 0; k0 < 64; k0 += 32) {
    const v16h a = load_afrag(&satt[(wave * 16 + l15) * 64], k0, half);
#pragma unroll
    for (int c = 0; c < 2; ++c) {
      const v16h b =
          *reinterpret_cast<const v16h*>(vb + (size_t)(c * 16 + l15) * TT + k0 + half * 16);
      acco[c] = __builtin_amdgcn_wmma_f32_16x16x32_f16(false, a, false, b,
                                                       (short)0, acco[c], false, false);
    }
  }
#pragma unroll
  for (int c = 0; c < 2; ++c) {
    const int col = c * 16 + l15;
#pragma unroll
    for (int r = 0; r < 8; ++r) {
      const int orow = wave * 16 + 8 * half + r;
      o[((size_t)n * TT + orow) * HH + head * DHD + col] = (_Float16)acco[c][r];
    }
  }
}

// ---------------- LayerNorm (wave per row), optional residual, f32+f16 out --

__global__ void __launch_bounds__(256) ln_kernel(
    const float* __restrict__ X, const float* __restrict__ R,
    const float* __restrict__ g, const float* __restrict__ b,
    float* __restrict__ Yf, _Float16* __restrict__ Yh, int M, int D) {
  const int row = blockIdx.x * 8 + (threadIdx.x >> 5);
  const int lane = threadIdx.x & 31;
  if (row >= M) return;
  const float* xr = X + (size_t)row * D;
  const float* rr = R ? R + (size_t)row * D : nullptr;
  float vals[12];          // D <= 384
  const int cnt = D >> 5;  // D is a multiple of 32 here (32/128/384)
  float s = 0.f, ss = 0.f;
  for (int i = 0; i < cnt; ++i) {
    float v = xr[lane + i * 32];
    if (rr) v += rr[lane + i * 32];
    vals[i] = v; s += v; ss += v * v;
  }
  for (int off = 16; off >= 1; off >>= 1) {
    s += __shfl_xor(s, off, 32);
    ss += __shfl_xor(ss, off, 32);
  }
  const float mu = s / (float)D;
  const float var = ss / (float)D - mu * mu;
  const float rstd = rsqrtf(var + 1e-5f);
  for (int i = 0; i < cnt; ++i) {
    const int c = lane + i * 32;
    const float y = (vals[i] - mu) * rstd * g[c] + b[c];
    if (Yf) Yf[(size_t)row * D + c] = y;
    if (Yh) Yh[(size_t)row * D + c] = (_Float16)y;
  }
}

// ---------------- attention pooling over T ----------------------------------

__global__ void __launch_bounds__(128) pool_kernel(
    const _Float16* __restrict__ tp, const float* __restrict__ x32,
    const float* __restrict__ pw2, const float* __restrict__ pb2,
    float* __restrict__ hd) {
  const int n = blockIdx.x;
  __shared__ float sc[TT];
  __shared__ float aw[TT];
  const int tid = threadIdx.x;
  if (tid < TT) {
    float acc = pb2[0];
    const _Float16* r = tp + ((size_t)n * TT + tid) * HH;
    for (int i = 0; i < HH; ++i) acc += (float)r[i] * pw2[i];
    sc[tid] = acc;
  }
  __syncthreads();
  if (tid == 0) {
    float m = -1e30f;
    for (int t = 0; t < TT; ++t) m = fmaxf(m, sc[t]);
    float sum = 0.f;
    for (int t = 0; t < TT; ++t) { aw[t] = expf(sc[t] - m); sum += aw[t]; }
    const float inv = 1.f / sum;
    for (int t = 0; t < TT; ++t) aw[t] *= inv;
  }
  __syncthreads();
  float acc = 0.f;
  for (int t = 0; t < TT; ++t) acc += aw[t] * x32[((size_t)n * TT + t) * HH + tid];
  hd[(size_t)n * HH + tid] = acc;
}

// ---------------- chain gather / gate-mix / FiLM ----------------------------

__global__ void chain_gather_kernel(const int* __restrict__ cid,
                                    const float* __restrict__ emb,
                                    _Float16* __restrict__ out) {
  const int n = blockIdx.x, j = threadIdx.x; // 32 threads
  const int c = clampi(cid[n], NCHAIN);
  out[(size_t)n * CEC + j] = (_Float16)emb[(size_t)c * CEC + j];
}

__global__ void __launch_bounds__(128) gate_mix_kernel(
    const float* __restrict__ hd, const float* __restrict__ hch,
    const float* __restrict__ gw, const float* __restrict__ gb,
    float* __restrict__ hp) {
  const int n = blockIdx.x, tid = threadIdx.x;
  __shared__ float red[128];
  red[tid] = gw[tid] * hd[(size_t)n * HH + tid] + gw[HH + tid] * hch[(size_t)n * HH + tid];
  __syncthreads();
  for (int s = 64; s > 0; s >>= 1) {
    if (tid < s) red[tid] += red[tid + s];
    __syncthreads();
  }
  const float g = 1.f / (1.f + expf(-(red[0] + gb[0])));
  hp[(size_t)n * HH + tid] =
      g * hd[(size_t)n * HH + tid] + (1.f - g) * hch[(size_t)n * HH + tid];
}

__global__ void film_kernel(const float* __restrict__ hp, const float* __restrict__ gam,
                            const float* __restrict__ bet, float* __restrict__ hc32,
                            _Float16* __restrict__ hc16) {
  const size_t i = (size_t)blockIdx.x * 256 + threadIdx.x;
  if (i >= (size_t)NN * HH) return;
  const float v = hp[i] * (1.f + gam[i]) + bet[i];
  hc32[i] = v;
  hc16[i] = (_Float16)v;
}

// ---------------- GAT: f/g projections + Wh^T; dense masked softmax ---------

__global__ void __launch_bounds__(256) gat_fg_kernel(
    const float* __restrict__ Wh, const float* __restrict__ att,
    float* __restrict__ f, float* __restrict__ g, _Float16* __restrict__ WhT) {
  const int row = blockIdx.x * 8 + (threadIdx.x >> 5);
  const int lane = threadIdx.x & 31;
  float sf = 0.f, sg = 0.f;
  float v[4];
#pragma unroll
  for (int i = 0; i < 4; ++i) {
    const int c = lane + i * 32;
    const float w = Wh[(size_t)row * HH + c];
    v[i] = w;
    sf += w * att[c];
    sg += w * att[HH + c];
  }
  for (int off = 16; off >= 1; off >>= 1) {
    sf += __shfl_xor(sf, off, 32);
    sg += __shfl_xor(sg, off, 32);
  }
  if (lane == 0) { f[row] = sf; g[row] = sg; }
#pragma unroll
  for (int i = 0; i < 4; ++i)
    WhT[(size_t)(lane + i * 32) * NN + row] = (_Float16)v[i];
}

__global__ void __launch_bounds__(256) gat_alpha_kernel(
    const float* __restrict__ f, const float* __restrict__ g,
    const float* __restrict__ att, const int* __restrict__ ind,
    _Float16* __restrict__ alpha, int clipmax) {
  const int i = blockIdx.x, tid = threadIdx.x;
  __shared__ float red[256];
  const int ii = clampi(ind[i], clipmax);
  const float fi = f[i], a3 = att[2 * HH];
  float m = -1e30f;
  for (int j = tid; j < NN; j += 256) {
    if (clampi(ind[j], clipmax) == ii) {
      float e = fi + g[j] + a3;
      e = e > 0.f ? e : 0.1f * e;
      m = fmaxf(m, e);
    }
  }
  red[tid] = m; __syncthreads();
  for (int s = 128; s > 0; s >>= 1) {
    if (tid < s) red[tid] = fmaxf(red[tid], red[tid + s]);
    __syncthreads();
  }
  m = red[0]; __syncthreads();
  float sum = 0.f;
  for (int j = tid; j < NN; j += 256) {
    if (clampi(ind[j], clipmax) == ii) {
      float e = fi + g[j] + a3;
      e = e > 0.f ? e : 0.1f * e;
      sum += expf(e - m);
    }
  }
  red[tid] = sum; __syncthreads();
  for (int s = 128; s > 0; s >>= 1) {
    if (tid < s) red[tid] += red[tid + s];
    __syncthreads();
  }
  const float inv = 1.f / red[0];
  for (int j = tid; j < NN; j += 256) {
    float v = 0.f;
    if (clampi(ind[j], clipmax) == ii) {
      float e = fi + g[j] + a3;
      e = e > 0.f ? e : 0.1f * e;
      v = expf(e - m) * inv;
    }
    alpha[(size_t)i * NN + j] = (_Float16)v;
  }
}

// ---------------- fuse / head glue ------------------------------------------

__global__ void pack2_kernel(const float* __restrict__ h2, const float* __restrict__ h1,
                             _Float16* __restrict__ out) {
  const size_t i = (size_t)blockIdx.x * 256 + threadIdx.x;
  if (i >= (size_t)NN * 2 * HH) return;
  const int col = (int)(i & 255);
  const size_t n = i >> 8;
  const float v = col < HH ? h2[n * HH + col] : h1[n * HH + col - HH];
  out[i] = (_Float16)v;
}

__global__ void fuse_mix_kernel(const float* __restrict__ g12, const float* __restrict__ h2,
                                const float* __restrict__ h1, float* __restrict__ out) {
  const size_t i = (size_t)blockIdx.x * 256 + threadIdx.x;
  if (i >= (size_t)NN * HH) return;
  const float g = g12[i];
  out[i] = g * h2[i] + (1.f - g) * h1[i];
}

__global__ void pack3_kernel(const float* __restrict__ a, const float* __restrict__ b,
                             const float* __restrict__ c, float* __restrict__ out) {
  const size_t i = (size_t)blockIdx.x * 256 + threadIdx.x;
  if (i >= (size_t)NN * 3 * HH) return;
  const int col = (int)(i % (3 * HH));
  const size_t n = i / (3 * HH);
  float v;
  if (col < HH) v = a[n * HH + col];
  else if (col < 2 * HH) v = b[n * HH + col - HH];
  else v = c[n * HH + col - 2 * HH];
  out[i] = v;
}

__global__ void __launch_bounds__(64) head_final_kernel(
    const _Float16* __restrict__ m2, const float* __restrict__ hw3,
    const float* __restrict__ hb3, float* __restrict__ out) {
  const int n = blockIdx.x, tid = threadIdx.x;
  __shared__ float red[64];
  red[tid] = (float)m2[(size_t)n * 64 + tid] * hw3[tid];
  __syncthreads();
  for (int s = 32; s > 0; s >>= 1) {
    if (tid < s) red[tid] += red[tid + s];
    __syncthreads();
  }
  if (tid == 0) out[n] = red[0] + hb3[0];
}

__global__ void cast16_kernel(const float* __restrict__ s, _Float16* __restrict__ d, int n) {
  const int i = blockIdx.x * 256 + threadIdx.x;
  if (i < n) d[i] = (_Float16)s[i];
}

// ---------------------------------------------------------------------------
// Host launcher.
// d_in flattening (setup_inputs() dict order; params in insertion order):
//  0 daily_feat[N,T,16]  1 chain_id[N]  2 ind1_id[N]  3 ind2_id[N]  4 ctx_feat[N,32]
//  5 conv_w 6 conv_b | tr[0]: 7 in_w 8 in_b 9 out_w 10 out_b 11 ln1_g 12 ln1_b
//  13 w1 14 b1 15 w2 16 b2 17 ln2_g 18 ln2_b | 19 pool_w1 20 pool_b1 21 pool_w2
//  22 pool_b2 23 chain_emb 24 chain_proj_w 25 chain_proj_b 26 gate_w 27 gate_b
//  28 ctx_norm_g 29 ctx_norm_b 30 film_gamma_w 31 film_gamma_b 32 film_beta_w
//  33 film_beta_b | g2[0]:34 W 35 att 36 ln_g 37 ln_b  g2[1]:38..41
//  g1[0]:42..45  g1[1]:46..49 | 50 fuse_w 51 fuse_b 52 fuse_norm_g 53 fuse_norm_b
//  54 head_ln_g 55 head_ln_b 56 hw1 57 hb1 58 hw2 59 hb2 60 hw3 61 hb3
// ---------------------------------------------------------------------------

static inline void cast16(hipStream_t s, const float* src, _Float16* dst, int n) {
  cast16_kernel<<<(n + 255) / 256, 256, 0, s>>>(src, dst, n);
}

extern "C" void kernel_launch(void* const* d_in, const int* in_sizes, int n_in,
                              void* d_out, int out_size, void* d_ws, size_t ws_size,
                              hipStream_t stream) {
  (void)in_sizes; (void)n_in; (void)out_size; (void)ws_size;

  const float* daily = (const float*)d_in[0];
  const int* chain_id = (const int*)d_in[1];
  const int* ind1 = (const int*)d_in[2];
  const int* ind2 = (const int*)d_in[3];
  const float* ctx = (const float*)d_in[4];
  const float* conv_w = (const float*)d_in[5];
  const float* conv_b = (const float*)d_in[6];
  const float* in_w = (const float*)d_in[7];
  const float* in_b = (const float*)d_in[8];
  const float* out_w = (const float*)d_in[9];
  const float* out_b = (const float*)d_in[10];
  const float* ln1_g = (const float*)d_in[11];
  const float* ln1_b = (const float*)d_in[12];
  const float* w1 = (const float*)d_in[13];
  const float* b1 = (const float*)d_in[14];
  const float* w2 = (const float*)d_in[15];
  const float* b2 = (const float*)d_in[16];
  const float* ln2_g = (const float*)d_in[17];
  const float* ln2_b = (const float*)d_in[18];
  const float* pool_w1 = (const float*)d_in[19];
  const float* pool_b1 = (const float*)d_in[20];
  const float* pool_w2 = (const float*)d_in[21];
  const float* pool_b2 = (const float*)d_in[22];
  const float* chain_emb = (const float*)d_in[23];
  const float* chain_proj_w = (const float*)d_in[24];
  const float* chain_proj_b = (const float*)d_in[25];
  const float* gate_w = (const float*)d_in[26];
  const float* gate_b = (const float*)d_in[27];
  const float* ctx_norm_g = (const float*)d_in[28];
  const float* ctx_norm_b = (const float*)d_in[29];
  const float* film_gamma_w = (const float*)d_in[30];
  const float* film_gamma_b = (const float*)d_in[31];
  const float* film_beta_w = (const float*)d_in[32];
  const float* film_beta_b = (const float*)d_in[33];
  const float* gW[4], *gatt[4], *glng[4], *glnb[4]; // [g2_0, g2_1, g1_0, g1_1]
  for (int l = 0; l < 4; ++l) {
    gW[l]   = (const float*)d_in[34 + 4 * l + 0];
    gatt[l] = (const float*)d_in[34 + 4 * l + 1];
    glng[l] = (const float*)d_in[34 + 4 * l + 2];
    glnb[l] = (const float*)d_in[34 + 4 * l + 3];
  }
  const float* fuse_w = (const float*)d_in[50];
  const float* fuse_b = (const float*)d_in[51];
  const float* fuse_norm_g = (const float*)d_in[52];
  const float* fuse_norm_b = (const float*)d_in[53];
  const float* head_ln_g = (const float*)d_in[54];
  const float* head_ln_b = (const float*)d_in[55];
  const float* hw1 = (const float*)d_in[56];
  const float* hb1 = (const float*)d_in[57];
  const float* hw2 = (const float*)d_in[58];
  const float* hb2 = (const float*)d_in[59];
  const float* hw3 = (const float*)d_in[60];
  const float* hb3 = (const float*)d_in[61];
  float* out = (float*)d_out;

  // ---- workspace bump allocation (deterministic every call) ----
  char* wp = (char*)d_ws;
  auto alloc = [&](size_t bytes) -> char* {
    char* r = wp;
    wp += (bytes + 255) & ~(size_t)255;
    return r;
  };
  const size_t MT = (size_t)NN * TT; // 131072 rows
  float*    x32    = (float*)alloc(MT * HH * 4);
  _Float16* xf16   = (_Float16*)alloc(MT * HH * 2);
  _Float16* qkv16  = (_Float16*)alloc(MT * 3 * HH * 2);
  _Float16* q16    = (_Float16*)alloc(MT * HH * 2);
  _Float16* k16    = (_Float16*)alloc(MT * HH * 2);
  _Float16* vT16   = (_Float16*)alloc(MT * HH * 2);
  _Float16* o16    = (_Float16*)alloc(MT * HH * 2); // reused as tanh-pool feats
  float*    tmp32  = (float*)alloc(MT * HH * 4);    // generic f32 GEMM output
  _Float16* h1_16  = (_Float16*)alloc(MT * 4 * HH * 2);
  _Float16* alpha16 = (_Float16*)alloc((size_t)NN * NN * 2);
  _Float16* WhT16  = (_Float16*)alloc((size_t)HH * NN * 2);
  float*    Wh32   = (float*)alloc((size_t)NN * HH * 4);
  float*    fbuf   = (float*)alloc((size_t)NN * 4);
  float*    gbuf   = (float*)alloc((size_t)NN * 4);
  float*    hd32   = (float*)alloc((size_t)NN * HH * 4);
  _Float16* cemb16 = (_Float16*)alloc((size_t)NN * CEC * 2);
  float*    hch32  = (float*)alloc((size_t)NN * HH * 4);
  float*    hp32   = (float*)alloc((size_t)NN * HH * 4);
  _Float16* ctxn16 = (_Float16*)alloc((size_t)NN * CTXC * 2);
  float*    gam32  = (float*)alloc((size_t)NN * HH * 4);
  float*    bet32  = (float*)alloc((size_t)NN * HH * 4);
  float*    hctx32 = (float*)alloc((size_t)NN * HH * 4);
  _Float16* hctx16 = (_Float16*)alloc((size_t)NN * HH * 2);
  float*    hg2_32 = (float*)alloc((size_t)NN * HH * 4);
  _Float16* hg2_16 = (_Float16*)alloc((size_t)NN * HH * 2);
  float*    hg1_32 = (float*)alloc((size_t)NN * HH * 4);
  _Float16* hg1_16 = (_Float16*)alloc((size_t)NN * HH * 2);
  _Float16* cat16  = (_Float16*)alloc((size_t)NN * 2 * HH * 2);
  float*    g12_32 = (float*)alloc((size_t)NN * HH * 4);
  float*    hgr32  = (float*)alloc((size_t)NN * HH * 4);
  float*    cat3_32 = (float*)alloc((size_t)NN * 3 * HH * 4);
  _Float16* cat3_16 = (_Float16*)alloc((size_t)NN * 3 * HH * 2);
  _Float16* m1_16  = (_Float16*)alloc((size_t)NN * 192 * 2);
  _Float16* m2_16  = (_Float16*)alloc((size_t)NN * 64 * 2);
  // f16 weights
  _Float16* in_w16   = (_Float16*)alloc(3 * HH * HH * 2);
  _Float16* out_w16  = (_Float16*)alloc(HH * HH * 2);
  _Float16* w1_16    = (_Float16*)alloc(4 * HH * HH * 2);
  _Float16* w2_16    = (_Float16*)alloc(HH * 4 * HH * 2);
  _Float16* pw1_16   = (_Float16*)alloc(HH * HH * 2);
  _Float16* cpw16    = (_Float16*)alloc(HH * CEC * 2);
  _Float16* fgw16    = (_Float16*)alloc(HH * CTXC * 2);
  _Float16* fbw16    = (_Float16*)alloc(HH * CTXC * 2);
  _Float16* gW16[4];
  for (int l = 0; l < 4; ++l) gW16[l] = (_Float16*)alloc(HH * HH * 2);
  _Float16* fuse16   = (_Float16*)alloc(HH * 2 * HH * 2);
  _Float16* hw1_16   = (_Float16*)alloc(192 * 3 * HH * 2);
  _Float16* hw2_16   = (_Float16*)alloc(64 * 192 * 2);

  // ---- weight casts ----
  cast16(stream, in_w, in_w16, 3 * HH * HH);
  cast16(stream, out_w, out_w16, HH * HH);
  cast16(stream, w1, w1_16, 4 * HH * HH);
  cast16(stream, w2, w2_16, HH * 4 * HH);
  cast16(stream, pool_w1, pw1_16, HH * HH);
  cast16(stream, chain_proj_w, cpw16, HH * CEC);
  cast16(stream, film_gamma_w, fgw16, HH * CTXC);
  cast16(stream, film_beta_w, fbw16, HH * CTXC);
  for (int l = 0; l < 4; ++l) cast16(stream, gW[l], gW16[l], HH * HH);
  cast16(stream, fuse_w, fuse16, HH * 2 * HH);
  cast16(stream, hw1, hw1_16, 192 * 3 * HH);
  cast16(stream, hw2, hw2_16, 64 * 192);

  #define GEMM(ACT, O16, A, W, BIAS, CF, CH, M, N, K)                                   \
    gemm_wmma_kernel<ACT, O16><<<dim3((M) / 128, (N) / 64), 128, 0, stream>>>(          \
        A, W, BIAS, CF, CH, M, N, K)

  // ---- DailyEncoder ----
  conv_glu_kernel<<<NN * TT, 128, 0, stream>>>(daily, conv_w, conv_b, x32, xf16);
  GEMM(ACT_NONE, true, xf16, in_w16, in_b, nullptr, qkv16, (int)MT, 3 * HH, HH);
  split_qkv_kernel<<<(int)((MT * HH + 255) / 256), 256, 0, stream>>>(qkv16, q16, k16, vT16);
  attn_kernel<<<NN * NHD, 128, 0, stream>>>(q16, k16, vT16, o16);
  GEMM(ACT_NONE, false, o16, out_w16, out_b, tmp32, nullptr, (int)MT, HH, HH);
  ln_kernel<<<(int)(MT / 8), 256, 0, stream>>>(tmp32, x32, ln1_g, ln1_b, x32, xf16, (int)MT, HH);
  GEMM(ACT_RELU, true, xf16, w1_16, b1, nullptr, h1_16, (int)MT, 4 * HH, HH);
  GEMM(ACT_NONE, false, h1_16, w2_16, b2, tmp32, nullptr, (int)MT, HH, 4 * HH);
  ln_kernel<<<(int)(MT / 8), 256, 0, stream>>>(tmp32, x32, ln2_g, ln2_b, x32, xf16, (int)MT, HH);
  GEMM(ACT_TANH, true, xf16, pw1_16, pool_b1, nullptr, o16, (int)MT, HH, HH);
  pool_kernel<<<NN, 128, 0, stream>>>(o16, x32, pool_w2, pool_b2, hd32);

  // ---- chain gate + FiLM ----
  chain_gather_kernel<<<NN, CEC, 0, stream>>>(chain_id, chain_emb, cemb16);
  GEMM(ACT_NONE, false, cemb16, cpw16, chain_proj_b, hch32, nullptr, NN, HH, CEC);
  gate_mix_kernel<<<NN, 128, 0, stream>>>(hd32, hch32, gate_w, gate_b, hp32);
  ln_kernel<<<NN / 8, 256, 0, stream>>>(ctx, nullptr, ctx_norm_g, ctx_norm_b,
                                        nullptr, ctxn16, NN, CTXC);
  GEMM(ACT_NONE, false, ctxn16, fgw16, film_gamma_b, gam32, nullptr, NN, HH, CTXC);
  GEMM(ACT_NONE, false, ctxn16, fbw16, film_beta_b, bet32, nullptr, NN, HH, CTXC);
  film_kernel<<<(NN * HH + 255) / 256, 256, 0, stream>>>(hp32, gam32, bet32, hctx32, hctx16);

  // ---- graph blocks (g2 uses ind2, g1 uses ind1) ----
  struct { const int* ind; int clip; float* h32; _Float16* h16; int l0, l1; } blocks[2] = {
      {ind2, NIND2, hg2_32, hg2_16, 0, 1},
      {ind1, NIND1, hg1_32, hg1_16, 2, 3}};
  for (int bidx = 0; bidx < 2; ++bidx) {
    const float* cur32 = hctx32;
    const _Float16* cur16 = hctx16;
    for (int li = 0; li < 2; ++li) {
      const int l = (li == 0) ? blocks[bidx].l0 : blocks[bidx].l1;
      GEMM(ACT_NONE, false, cur16, gW16[l], nullptr, Wh32, nullptr, NN, HH, HH);
      gat_fg_kernel<<<NN / 8, 256, 0, stream>>>(Wh32, gatt[l], fbuf, gbuf, WhT16);
      gat_alpha_kernel<<<NN, 256, 0, stream>>>(fbuf, gbuf, gatt[l], blocks[bidx].ind,
                                               alpha16, blocks[bidx].clip);
      GEMM(ACT_NONE, false, alpha16, WhT16, nullptr, tmp32, nullptr, NN, HH, NN);
      ln_kernel<<<NN / 8, 256, 0, stream>>>(tmp32, cur32, glng[l], glnb[l],
                                            blocks[bidx].h32, blocks[bidx].h16, NN, HH);
      cur32 = blocks[bidx].h32;
      cur16 = blocks[bidx].h16;
    }
  }

  // ---- fuse ----
  pack2_kernel<<<(NN * 2 * HH + 255) / 256, 256, 0, stream>>>(hg2_32, hg1_32, cat16);
  GEMM(ACT_SIGMOID, false, cat16, fuse16, fuse_b, g12_32, nullptr, NN, HH, 2 * HH);
  fuse_mix_kernel<<<(NN * HH + 255) / 256, 256, 0, stream>>>(g12_32, hg2_32, hg1_32, tmp32);
  ln_kernel<<<NN / 8, 256, 0, stream>>>(tmp32, nullptr, fuse_norm_g, fuse_norm_b,
                                        hgr32, nullptr, NN, HH);

  // ---- head ----
  pack3_kernel<<<(NN * 3 * HH + 255) / 256, 256, 0, stream>>>(hp32, hctx32, hgr32, cat3_32);
  ln_kernel<<<NN / 8, 256, 0, stream>>>(cat3_32, nullptr, head_ln_g, head_ln_b,
                                        nullptr, cat3_16, NN, 3 * HH);
  GEMM(ACT_SILU, true, cat3_16, hw1_16, hb1, nullptr, m1_16, NN, 192, 3 * HH);
  GEMM(ACT_SILU, true, m1_16, hw2_16, hb2, nullptr, m2_16, NN, 64, 192);
  head_final_kernel<<<NN, 64, 0, stream>>>(m2_16, hw3, hb3, out);
  #undef GEMM
}
